// EquivMultiHeadQueryModel_4114578670367
// MI455X (gfx1250) — compile-verified, matrix operand order checked by reading, not run on hardware
//
#include <hip/hip_runtime.h>
#include <math.h>

// ---------------- model dims ----------------
#define NN   4096          // nodes
#define NE   49152         // edges
#define NQn  8             // queries
#define NPp  (NN*NQn)      // query-node pairs = 32768
#define Cc   32            // channels
#define Mm   16            // SH components
#define Bb   10            // RBF basis
#define Hh   32            // MLP hidden
#define XST  (Cc*Mm)       // 512 floats per node feature block

typedef __attribute__((ext_vector_type(16))) _Float16 v16h;
typedef __attribute__((ext_vector_type(8)))  float    v8f;

// ---------------- fast transcendentals (v_exp_f32 / v_rcp_f32) ------------
__device__ __forceinline__ float fexp(float x) { return __expf(x); }
__device__ __forceinline__ float frcp(float x) { return __builtin_amdgcn_rcpf(x); }
__device__ __forceinline__ float fsilu(float x) {
  return x * frcp(1.0f + __expf(-x));
}

// ---------------- WMMA fragment helpers (ISA 7.12.2, 16-bit A/B 16x32) ------
// element e (0..15) of the v16h per-lane fragment; hi = lane>=16
__device__ __forceinline__ int wmma_k(int e, int hi) {
  int v = e >> 1;
  return ((v & 4) << 2) + (hi << 3) + ((v & 3) << 1) + (e & 1);
}
// inverse: for K index k, which (lane-half,e) holds it
__device__ __forceinline__ void wmma_ke_inv(int k, int* dhi, int* e) {
  int A = (k >> 4) & 1, h = (k >> 3) & 1, b = (k >> 1) & 3, d = k & 1;
  *dhi = h;
  *e = ((A << 2) | b) * 2 + d;
}

__device__ __forceinline__ float wsum32(float v) {
  #pragma unroll
  for (int off = 16; off > 0; off >>= 1) v += __shfl_xor(v, off, 32);
  return v;
}

__device__ __forceinline__ int lofm(int m) {
  return (m == 0) ? 0 : (m < 4) ? 1 : (m < 9) ? 2 : 3;
}

// ordered-uint encoding for float atomicMax
__device__ __forceinline__ unsigned enc_ord(float f) {
  unsigned u = __float_as_uint(f);
  return (u & 0x80000000u) ? ~u : (u | 0x80000000u);
}
__device__ __forceinline__ float dec_ord(unsigned u) {
  return __uint_as_float((u & 0x80000000u) ? (u & 0x7FFFFFFFu) : ~u);
}

// ---------------- B-operand pre-pack into WMMA fragment layout -------------
// dst[tile*512 + lane*16 + e] = W[k(e,hi)*CO + tile*16 + (lane&15)], zero-padded
__global__ __launch_bounds__(256) void k_packB(
    const float* __restrict__ W, _Float16* __restrict__ dst, int K, int CO) {
  int ntile = (CO + 15) >> 4;
  int t = blockIdx.x * blockDim.x + threadIdx.x;
  if (t >= ntile * 512) return;
  int tile = t >> 9, r = t & 511;
  int lane = r >> 4, e = r & 15;
  int hi = lane >> 4, lr = lane & 15;
  int o = tile * 16 + lr;
  int k = wmma_k(e, hi);
  float v = (k < K && o < CO) ? W[k * CO + o] : 0.0f;
  dst[(size_t)tile * 512 + lane * 16 + e] = (_Float16)v;
}

// ---------------- geometry: sh16 / rbf / cutoff ----------------
__device__ void geom_one(float rx, float ry, float rz, float rmax,
                         float* sh, float* rb, float* wout) {
  float d = sqrtf(rx * rx + ry * ry + rz * rz + 1e-12f);
  float inv = frcp(d + 1e-9f);
  float ux = rx * inv, uy = ry * inv, uz = rz * inv;
  sh[0] = 1.0f;
  float n1 = frcp(sqrtf(ux * ux + uy * uy + uz * uz) + 1e-9f);
  sh[1] = ux * n1; sh[2] = uy * n1; sh[3] = uz * n1;
  const float s15 = 3.8729833f;
  float t[7];
  t[0] = s15 * ux * uy;
  t[1] = s15 * uy * uz;
  t[2] = 1.1180340f * (3.f * uz * uz - 1.f);
  t[3] = s15 * ux * uz;
  t[4] = 0.5f * s15 * (ux * ux - uy * uy);
  float n2 = 0.f;
  #pragma unroll
  for (int i = 0; i < 5; ++i) n2 += t[i] * t[i];
  n2 = frcp(sqrtf(n2) + 1e-9f);
  #pragma unroll
  for (int i = 0; i < 5; ++i) sh[4 + i] = t[i] * n2;
  const float a = 1.0458250f, b = 5.1234753f, c = 0.8100925f, dd = 0.6614378f;
  t[0] = a * uy * (3.f * ux * ux - uy * uy);
  t[1] = b * ux * uy * uz;
  t[2] = c * uy * (5.f * uz * uz - 1.f);
  t[3] = dd * uz * (5.f * uz * uz - 3.f);
  t[4] = c * ux * (5.f * uz * uz - 1.f);
  t[5] = 0.5f * b * uz * (ux * ux - uy * uy);
  t[6] = a * ux * (ux * ux - 3.f * uy * uy);
  float n3 = 0.f;
  #pragma unroll
  for (int i = 0; i < 7; ++i) n3 += t[i] * t[i];
  n3 = frcp(sqrtf(n3) + 1e-9f);
  #pragma unroll
  for (int i = 0; i < 7; ++i) sh[9 + i] = t[i] * n3;
  float step = rmax / 9.0f, width = rmax / (float)Bb;
  float invw = frcp(width);
  #pragma unroll
  for (int i = 0; i < Bb; ++i) {
    float z = (d - step * (float)i) * invw;
    rb[i] = fexp(-z * z);
  }
  *wout = (d < rmax) ? 0.5f * (__cosf(3.14159265358979f * d / rmax) + 1.0f) : 0.0f;
}

__global__ __launch_bounds__(256) void k_geom_edges(
    const float* __restrict__ pos, const int* __restrict__ src,
    const int* __restrict__ dst, float rmax,
    float* __restrict__ sh, float* __restrict__ rb, float* __restrict__ w, int nE) {
  int e = blockIdx.x * blockDim.x + threadIdx.x;
  if (e >= nE) return;
  int s = src[e], d = dst[e];
  geom_one(pos[d * 3 + 0] - pos[s * 3 + 0],
           pos[d * 3 + 1] - pos[s * 3 + 1],
           pos[d * 3 + 2] - pos[s * 3 + 2],
           rmax, sh + (size_t)e * 16, rb + (size_t)e * Bb, w + e);
}

__global__ __launch_bounds__(256) void k_geom_pairs(
    const float* __restrict__ pos, const float* __restrict__ qp, float rmax,
    float* __restrict__ sh, float* __restrict__ rb, float* __restrict__ w, int nP) {
  int p = blockIdx.x * blockDim.x + threadIdx.x;
  if (p >= nP) return;
  int s = p % NN, q = p / NN;
  geom_one(qp[q * 3 + 0] - pos[s * 3 + 0],
           qp[q * 3 + 1] - pos[s * 3 + 1],
           qp[q * 3 + 2] - pos[s * 3 + 2],
           rmax, sh + (size_t)p * 16, rb + (size_t)p * Bb, w + p);
}

// ---------------- x seed: x[:, :, 0] = feature @ emb ----------------
__global__ __launch_bounds__(256) void k_seed_x(
    const float* __restrict__ feature, const float* __restrict__ emb,
    float* __restrict__ x) {
  int idx = blockIdx.x * blockDim.x + threadIdx.x;  // n*32+c
  if (idx >= NN * Cc) return;
  int n = idx >> 5, c = idx & 31;
  float v = feature[n * 3 + 0] * emb[0 * Cc + c] +
            feature[n * 3 + 1] * emb[1 * Cc + c] +
            feature[n * 3 + 2] * emb[2 * Cc + c];
  float* p = x + (size_t)n * XST + c * Mm;
  p[0] = v;
  #pragma unroll
  for (int m = 1; m < Mm; ++m) p[m] = 0.0f;
}

// ---------------- channel mixing GEMM via WMMA --------------------------
// rows = (n,m), K = 32 channels. One wave per node (16 m-rows).
// A tile staged global->LDS with async-to-LDS DMA, B pre-packed fragments.
template <int CO, int NMAT>
__global__ __launch_bounds__(256) void k_cmix(
    const float* __restrict__ X,                       // [nrows, 32, 16]
    const _Float16* __restrict__ B0, const _Float16* __restrict__ B1,
    const _Float16* __restrict__ B2, const _Float16* __restrict__ B3,
    float* __restrict__ out, int nrows) {
  alignas(16) __shared__ float tile[8][512];
  int w = threadIdx.x >> 5;
  int wid = blockIdx.x * 8 + w;
  if (wid >= nrows) return;
  int lane = threadIdx.x & 31;
  int hi = lane >> 4, lr = lane & 15;
  const float* xn = X + (size_t)wid * XST;
  // ---- stage 2KB node tile into LDS via async DMA (ASYNCcnt-tracked) ----
  unsigned ldsbase = (unsigned)(uintptr_t)&tile[w][0];
  unsigned long long gbase = (unsigned long long)(uintptr_t)xn;
  #pragma unroll
  for (int j = 0; j < 4; ++j) {
    unsigned la = ldsbase + (unsigned)(j * 512 + lane * 16);
    unsigned long long ga = gbase + (unsigned long long)(j * 512 + lane * 16);
    asm volatile("global_load_async_to_lds_b128 %0, %1, off"
                 :: "v"(la), "v"(ga) : "memory");
  }
  asm volatile("s_wait_asynccnt 0x0" ::: "memory");
  // ---- assemble A fragment from LDS ----
  v16h a;
  #pragma unroll
  for (int e = 0; e < 16; ++e) {
    int k = wmma_k(e, hi);
    a[e] = (_Float16)tile[w][k * Mm + lr];   // row = m = lr, K = channel
  }
  constexpr int NT = (CO + 15) / 16;
  #pragma unroll
  for (int wm = 0; wm < NMAT; ++wm) {
    const _Float16* pB = (wm == 0) ? B0 : (wm == 1) ? B1 : (wm == 2) ? B2 : B3;
    float* op = out + (size_t)wm * nrows * CO * Mm + (size_t)wid * CO * Mm;
    #pragma unroll
    for (int t = 0; t < NT; ++t) {
      v16h b = *(const v16h*)(pB + (size_t)t * 512 + lane * 16);
      v8f acc = {};
      acc = __builtin_amdgcn_wmma_f32_16x16x32_f16(false, a, false, b,
                                                   (short)0, acc, false, false);
      int o = t * 16 + lr;
      if ((CO & 15) == 0 || o < CO) {
        #pragma unroll
        for (int r = 0; r < 8; ++r) {
          int m = r + 8 * hi;
          op[o * Mm + m] = acc[r];
        }
      }
    }
  }
}

// ---------------- gate MLP: GEMM1 (rb @ W1, K=10 padded to 32) + silu ----
// writes hid directly in WMMA-A fragment layout: hidp[tile*512 + lane*16 + e]
__global__ __launch_bounds__(256) void k_mlp1(
    const float* __restrict__ rb, const _Float16* __restrict__ pW1,
    const float* __restrict__ b1, _Float16* __restrict__ hidp, int P) {
  int wid = blockIdx.x * (blockDim.x >> 5) + (threadIdx.x >> 5);
  if (wid >= (P >> 4)) return;
  int lane = threadIdx.x & 31;
  int hi = lane >> 4, lr = lane & 15;
  const float* rbrow = rb + (size_t)(wid * 16 + lr) * Bb;
  v16h a;
  #pragma unroll
  for (int e = 0; e < 16; ++e) {
    int k = wmma_k(e, hi);
    float v = rbrow[(k < Bb) ? k : 0];       // unconditional load, clamp addr
    a[e] = (k < Bb) ? (_Float16)v : (_Float16)0.0f;
  }
  #pragma unroll
  for (int blk = 0; blk < 2; ++blk) {
    int o = blk * 16 + lr;
    v16h b = *(const v16h*)(pW1 + (size_t)blk * 512 + lane * 16);
    v8f acc = {};
    acc = __builtin_amdgcn_wmma_f32_16x16x32_f16(false, a, false, b,
                                                 (short)0, acc, false, false);
    float bias = b1[o];
    int dhi, e;
    wmma_ke_inv(o, &dhi, &e);                // where col o lives as a K index
    #pragma unroll
    for (int r = 0; r < 8; ++r) {
      int rowloc = r + 8 * hi;               // tile-local pair row
      float s = fsilu(acc[r] + bias);
      int dlane = rowloc + 16 * dhi;
      hidp[(size_t)wid * 512 + dlane * 16 + e] = (_Float16)s;
    }
  }
}

// ---------------- gate MLP: GEMM2 (hid @ W2, K=32) + bias, f16 out -------
__global__ __launch_bounds__(256) void k_mlp2(
    const _Float16* __restrict__ hidp, const _Float16* __restrict__ pW2,
    const float* __restrict__ b2, _Float16* __restrict__ g, int P, int G) {
  int ct = G >> 4;
  int wid = blockIdx.x * (blockDim.x >> 5) + (threadIdx.x >> 5);
  if (wid >= (P >> 4) * ct) return;
  int rt_i = wid / ct, ct_i = wid % ct;
  int lane = threadIdx.x & 31;
  int hi = lane >> 4, lr = lane & 15;
  v16h a = *(const v16h*)(hidp + (size_t)rt_i * 512 + lane * 16);
  v16h b = *(const v16h*)(pW2 + (size_t)ct_i * 512 + lane * 16);
  v8f acc = {};
  acc = __builtin_amdgcn_wmma_f32_16x16x32_f16(false, a, false, b,
                                               (short)0, acc, false, false);
  int o = ct_i * 16 + lr;
  float bias = b2[o];
  #pragma unroll
  for (int r = 0; r < 8; ++r) {
    int prow = rt_i * 16 + r + 8 * hi;
    g[(size_t)prow * G + o] = (_Float16)(acc[r] + bias);
  }
}

// ---------------- attention logits (wave per edge, lane = channel) --------
__global__ __launch_bounds__(256) void k_logits(
    const float* __restrict__ q, const float* __restrict__ k0,
    const _Float16* __restrict__ g, const int* __restrict__ src,
    const int* __restrict__ dst, float* __restrict__ logits,
    unsigned* __restrict__ mbuf, int nE) {
  int e = blockIdx.x * (blockDim.x >> 5) + (threadIdx.x >> 5);
  if (e >= nE) return;
  int lane = threadIdx.x & 31;
  int s = src[e], d = dst[e];
  const float4* q4 = (const float4*)(q + (size_t)d * XST + lane * Mm);
  const float4* k4 = (const float4*)(k0 + (size_t)s * XST + lane * Mm);
  const _Float16* ge = g + (size_t)e * 384 + lane * 4;  // g[e,0,c,l]
  float gl[4];
  #pragma unroll
  for (int l = 0; l < 4; ++l) gl[l] = (float)ge[l];
  float qv[16], kv[16];
  #pragma unroll
  for (int j = 0; j < 4; ++j) {
    *(float4*)&qv[j * 4] = q4[j];
    *(float4*)&kv[j * 4] = k4[j];
  }
  float sum = 0.f;
  #pragma unroll
  for (int m = 0; m < Mm; ++m) sum += qv[m] * kv[m] * gl[lofm(m)];
  sum = wsum32(sum);
  if (lane == 0) {
    float lg = sum / 22.62741699796952f;   // sqrt(C*M)
    logits[e] = lg;
    atomicMax(&mbuf[d], enc_ord(lg));
  }
}

__global__ __launch_bounds__(256) void k_softden(
    const float* __restrict__ logits, const unsigned* __restrict__ mbuf,
    const int* __restrict__ dst, float* __restrict__ ealpha,
    float* __restrict__ den, int nE) {
  int e = blockIdx.x * blockDim.x + threadIdx.x;
  if (e >= nE) return;
  int d = dst[e];
  unsigned u = mbuf[d];
  float m = (u == 0u) ? 0.0f : dec_ord(u);
  float ev = fexp(logits[e] - m);
  ealpha[e] = ev;
  atomicAdd(&den[d], ev);
}

// ---------------- attention value aggregation (wave per edge) -------------
__global__ __launch_bounds__(256) void k_aggregate(
    const float* __restrict__ ealpha, const float* __restrict__ den,
    const float* __restrict__ wcut, const _Float16* __restrict__ g,
    const float* __restrict__ v0, const float* __restrict__ x,
    const float* __restrict__ sh, const int* __restrict__ src,
    const int* __restrict__ dst, float* __restrict__ agg, int nE) {
  int e = blockIdx.x * (blockDim.x >> 5) + (threadIdx.x >> 5);
  if (e >= nE) return;
  int lane = threadIdx.x & 31;
  int s = src[e], d = dst[e];
  float coeff = ealpha[e] * frcp(den[d] + 1e-9f) * wcut[e];
  const float4* v4 = (const float4*)(v0 + (size_t)s * XST + lane * Mm);
  const float4* s4 = (const float4*)(sh + (size_t)e * 16);
  float xs0 = x[(size_t)s * XST + lane * Mm];
  const _Float16* g1 = g + (size_t)e * 384 + 128 + lane * 4;  // g[e,1,c,l]
  const _Float16* g2 = g + (size_t)e * 384 + 256 + lane * 4;  // g[e,2,c,l]
  float g1l[4], g2l[4];
  #pragma unroll
  for (int l = 0; l < 4; ++l) { g1l[l] = (float)g1[l]; g2l[l] = (float)g2[l]; }
  float vv[16], she[16];
  #pragma unroll
  for (int j = 0; j < 4; ++j) {
    *(float4*)&vv[j * 4] = v4[j];
    *(float4*)&she[j * 4] = s4[j];
  }
  float* ad = agg + (size_t)d * XST + lane * Mm;
  #pragma unroll
  for (int m = 0; m < Mm; ++m) {
    int l = lofm(m);
    float v = vv[m] * g1l[l] + g2l[l] * xs0 * she[m];
    atomicAdd(&ad[m], coeff * v);
  }
}

// ---------------- skip + equivariant layernorm (wave per node) ------------
__global__ __launch_bounds__(256) void k_ln_x(
    const float* __restrict__ agg, const float* __restrict__ si,
    float* __restrict__ x, const float* __restrict__ gamma, int skip, int n) {
  int wid = blockIdx.x * (blockDim.x >> 5) + (threadIdx.x >> 5);
  if (wid >= n) return;
  int c = threadIdx.x & 31;
  size_t base = (size_t)wid * XST + c * Mm;
  float o[Mm];
  #pragma unroll
  for (int j = 0; j < 4; ++j) {
    float4 av = ((const float4*)(agg + base))[j];
    float4 sv = ((const float4*)(si + base))[j];
    o[j * 4 + 0] = av.x + sv.x;
    o[j * 4 + 1] = av.y + sv.y;
    o[j * 4 + 2] = av.z + sv.z;
    o[j * 4 + 3] = av.w + sv.w;
  }
  if (skip) {
    #pragma unroll
    for (int j = 0; j < 4; ++j) {
      float4 xv = ((const float4*)(x + base))[j];
      o[j * 4 + 0] += xv.x; o[j * 4 + 1] += xv.y;
      o[j * 4 + 2] += xv.z; o[j * 4 + 3] += xv.w;
    }
  }
  float bs[4] = {0.f, 0.f, 0.f, 0.f};
  #pragma unroll
  for (int m = 0; m < Mm; ++m) bs[lofm(m)] += o[m] * o[m];
  float rms[4];
  #pragma unroll
  for (int l = 0; l < 4; ++l)
    rms[l] = frcp(sqrtf(wsum32(bs[l]) * (1.0f / 32.0f) + 1e-6f));
  float gv[4];
  #pragma unroll
  for (int l = 0; l < 4; ++l) gv[l] = gamma[c * 4 + l] * rms[l];
  #pragma unroll
  for (int j = 0; j < 4; ++j) {
    float4 ov;
    ov.x = o[j * 4 + 0] * gv[lofm(j * 4 + 0)];
    ov.y = o[j * 4 + 1] * gv[lofm(j * 4 + 1)];
    ov.z = o[j * 4 + 2] * gv[lofm(j * 4 + 2)];
    ov.w = o[j * 4 + 3] * gv[lofm(j * 4 + 3)];
    ((float4*)(x + base))[j] = ov;
  }
}

// ---------------- field message + segment sum (wave per pair) -------------
// mode 0: s=src[p], q=dst[p];  mode 1: s=p%NN, q=p/NN
__global__ __launch_bounds__(256) void k_field_msg(
    int mode, const int* __restrict__ src, const int* __restrict__ dst,
    const float* __restrict__ xW, const _Float16* __restrict__ g,
    const float* __restrict__ sh, const float* __restrict__ wcut,
    float* __restrict__ num, float* __restrict__ den, int nP, int co) {
  int p = blockIdx.x * (blockDim.x >> 5) + (threadIdx.x >> 5);
  if (p >= nP) return;
  int lane = threadIdx.x & 31;
  int s, q;
  if (mode == 0) { s = src[p]; q = dst[p]; }
  else           { s = p % NN; q = p / NN; }
  float wv = wcut[p];
  if (lane < co) {
    const float4* xs4 = (const float4*)(xW + (size_t)s * co * Mm + lane * Mm);
    const float4* sh4 = (const float4*)(sh + (size_t)p * 16);
    int G = 8 * co;  // 2*co*4
    const _Float16* g0 = g + (size_t)p * G + lane * 4;
    const _Float16* g1 = g + (size_t)p * G + co * 4 + lane * 4;
    float g0l[4], g1l[4];
    #pragma unroll
    for (int l = 0; l < 4; ++l) { g0l[l] = (float)g0[l]; g1l[l] = (float)g1[l]; }
    float xs[16], shp[16];
    #pragma unroll
    for (int j = 0; j < 4; ++j) {
      *(float4*)&xs[j * 4] = xs4[j];
      *(float4*)&shp[j * 4] = sh4[j];
    }
    float xs0 = xs[0];
    float* np = num + (size_t)q * co * Mm + lane * Mm;
    #pragma unroll
    for (int m = 0; m < Mm; ++m) {
      int l = lofm(m);
      float msg = g0l[l] * xs[m] + g1l[l] * xs0 * shp[m];
      atomicAdd(&np[m], wv * msg);
    }
  }
  if (lane == 0) atomicAdd(&den[q], wv);
}

// ---------------- field finalize: num/(den+eps) + eq-layernorm ------------
__global__ __launch_bounds__(256) void k_field_fin(
    const float* __restrict__ num, const float* __restrict__ den,
    const float* __restrict__ gamma, float* __restrict__ out, int nq, int co) {
  int wid = blockIdx.x * (blockDim.x >> 5) + (threadIdx.x >> 5);
  if (wid >= nq) return;
  int c = threadIdx.x & 31;
  float invd = frcp(den[wid] + 1e-6f);
  float o[Mm];
  float bs[4] = {0.f, 0.f, 0.f, 0.f};
  size_t base = (size_t)wid * co * Mm + c * Mm;
  if (c < co) {
    #pragma unroll
    for (int j = 0; j < 4; ++j) {
      float4 nv = ((const float4*)(num + base))[j];
      o[j * 4 + 0] = nv.x * invd; o[j * 4 + 1] = nv.y * invd;
      o[j * 4 + 2] = nv.z * invd; o[j * 4 + 3] = nv.w * invd;
    }
    #pragma unroll
    for (int m = 0; m < Mm; ++m) bs[lofm(m)] += o[m] * o[m];
  } else {
    #pragma unroll
    for (int m = 0; m < Mm; ++m) o[m] = 0.f;
  }
  float rms[4];
  #pragma unroll
  for (int l = 0; l < 4; ++l)
    rms[l] = frcp(sqrtf(wsum32(bs[l]) * frcp((float)co) + 1e-6f));
  if (c < co) {
    #pragma unroll
    for (int m = 0; m < Mm; ++m) {
      int l = lofm(m);
      out[base + m] = o[m] * rms[l] * gamma[c * 4 + l];
    }
  }
}

// ---------------- pw head: softmax^(1/T) weights, accumulate qp sums ------
__global__ __launch_bounds__(256) void k_pw_head(
    const float* __restrict__ f, const float* __restrict__ pos,
    const float* __restrict__ headW, const float* __restrict__ headb,
    float* __restrict__ hbuf /* [0..23]=S, [24..31]=colsum */) {
  int n = blockIdx.x * blockDim.x + threadIdx.x;
  if (n >= NN) return;
  float lg[NQn];
  #pragma unroll
  for (int h = 0; h < NQn; ++h) lg[h] = headb[h];
  #pragma unroll
  for (int c = 0; c < 16; ++c) {
    float f0 = f[(size_t)n * 256 + c * Mm];      // f[n, c, 0], co=16
    #pragma unroll
    for (int h = 0; h < NQn; ++h) lg[h] += f0 * headW[c * NQn + h];
  }
  float mx = lg[0];
  #pragma unroll
  for (int h = 1; h < NQn; ++h) mx = fmaxf(mx, lg[h]);
  float s = 0.f, wq[NQn];
  #pragma unroll
  for (int h = 0; h < NQn; ++h) { wq[h] = fexp(lg[h] - mx); s += wq[h]; }
  float invs = frcp(s);
  #pragma unroll
  for (int h = 0; h < NQn; ++h) {
    float v = wq[h] * invs + 1e-4f;
    v = __powf(v, 33.333332f);   // 1/0.03
    wq[h] = v;
  }
  float px = pos[n * 3 + 0], py = pos[n * 3 + 1], pz = pos[n * 3 + 2];
  #pragma unroll
  for (int h = 0; h < NQn; ++h) {
    atomicAdd(&hbuf[24 + h], wq[h]);
    atomicAdd(&hbuf[h * 3 + 0], wq[h] * px);
    atomicAdd(&hbuf[h * 3 + 1], wq[h] * py);
    atomicAdd(&hbuf[h * 3 + 2], wq[h] * pz);
  }
}

__global__ void k_qp_init(const float* __restrict__ hbuf, float* __restrict__ qp) {
  int t = threadIdx.x;
  if (t >= NQn * 3) return;
  int q = t / 3;
  qp[t] = hbuf[t] / (hbuf[24 + q] + 1e-30f);
}

// qp[q,x] += sum_c fd[q, c, 1+x] * headW[c, q]   (diagonal of dall)
__global__ void k_disp_upd(const float* __restrict__ fd,
                           const float* __restrict__ headW,
                           float* __restrict__ qp) {
  int t = threadIdx.x;
  if (t >= NQn * 3) return;
  int q = t / 3, x = t % 3;
  float acc = 0.f;
  #pragma unroll
  for (int c = 0; c < 16; ++c)
    acc += fd[(size_t)q * 256 + c * Mm + 1 + x] * headW[c * NQn + q];
  qp[t] += acc;
}

// attn head: alog[q] = sum_c fa[q,c,0]*headW[c] + headb; softmax over q
__global__ void k_attn_head(const float* __restrict__ fa,
                            const float* __restrict__ headW,
                            const float* __restrict__ headb,
                            float* __restrict__ out) {
  if (threadIdx.x != 0) return;
  float lg[NQn];
  float mx = -1e30f;
  for (int q = 0; q < NQn; ++q) {
    float a = headb[0];
    for (int c = 0; c < 10; ++c) a += fa[(size_t)q * 160 + c * Mm] * headW[c];
    lg[q] = a;
    mx = fmaxf(mx, a);
  }
  float s = 0.f;
  for (int q = 0; q < NQn; ++q) { lg[q] = fexp(lg[q] - mx); s += lg[q]; }
  float invs = frcp(s);
  for (int q = 0; q < NQn; ++q) out[q] = lg[q] * invs;
}

// ================= host-side sequencing =================
static inline int divUp(int a, int b) { return (a + b - 1) / b; }

extern "C" void kernel_launch(void* const* d_in, const int* in_sizes, int n_in,
                              void* d_out_v, int out_size, void* d_ws, size_t ws_size,
                              hipStream_t stream) {
  (void)in_sizes; (void)n_in; (void)out_size; (void)ws_size;
  float* d_out = (float*)d_out_v;
  const float* feature = (const float*)d_in[0];
  const float* pos     = (const float*)d_in[1];
  const int*   esrc    = (const int*)d_in[2];
  const int*   edst    = (const int*)d_in[3];
  const float* TA[10]; for (int i = 0; i < 10; ++i) TA[i] = (const float*)d_in[4 + i];
  const float* TB[10]; for (int i = 0; i < 10; ++i) TB[i] = (const float*)d_in[14 + i];
  const float* PW[8];  for (int i = 0; i < 8;  ++i) PW[i] = (const float*)d_in[24 + i];
  const float* DP[8];  for (int i = 0; i < 8;  ++i) DP[i] = (const float*)d_in[32 + i];
  const float* FT[6];  for (int i = 0; i < 6;  ++i) FT[i] = (const float*)d_in[40 + i];
  const float* AT[8];  for (int i = 0; i < 8;  ++i) AT[i] = (const float*)d_in[46 + i];
  // TP idx: 0 emb 1 Wq 2 Wk 3 Wv 4 Wsi 5 W1 6 b1 7 W2 8 b2 9 gamma
  // FP idx: 0 Wf 1 W1 2 b1 3 W2 4 b2 5 gamma 6 headW 7 headb

  // --------- carve workspace ---------
  char* wsb = (char*)d_ws;
  size_t cur = 0;
  auto carve = [&](size_t bytes) -> void* {
    cur = (cur + 255) & ~(size_t)255;
    void* p = wsb + cur;
    cur += bytes;
    return p;
  };
  float*    sh_e  = (float*)carve((size_t)NE * 16 * 4);
  float*    rb_e  = (float*)carve((size_t)NE * Bb * 4);
  float*    w_e   = (float*)carve((size_t)NE * 4);
  float*    sh_p  = (float*)carve((size_t)NE * 16 * 4);
  float*    rb_p  = (float*)carve((size_t)NE * Bb * 4);
  float*    w_p   = (float*)carve((size_t)NE * 4);
  float*    xbuf  = (float*)carve((size_t)NN * XST * 4);
  float*    qkvs  = (float*)carve((size_t)4 * NN * XST * 4);
  float*    agg   = (float*)carve((size_t)NN * XST * 4);
  float*    feata = (float*)carve((size_t)NN * XST * 4);
  float*    featb = (float*)carve((size_t)NN * XST * 4);
  float*    xW    = (float*)carve((size_t)NN * XST * 4);
  _Float16* hidp  = (_Float16*)carve((size_t)NE * Hh * 2);
  _Float16* gbuf  = (_Float16*)carve((size_t)NE * 384 * 2);
  float*    logit = (float*)carve((size_t)NE * 4);
  float*    ealp  = (float*)carve((size_t)NE * 4);
  unsigned* mbuf  = (unsigned*)carve((size_t)NN * 4);
  float*    denb  = (float*)carve((size_t)NN * 4);
  float*    numf  = (float*)carve((size_t)NN * 256 * 4);
  float*    denf  = (float*)carve((size_t)NN * 4);
  float*    f_pw  = (float*)carve((size_t)NN * 256 * 4);
  float*    hbuf  = (float*)carve(64 * 4);
  float*    qp    = (float*)carve(32 * 4);
  float*    f_dsp = (float*)carve((size_t)NQn * 256 * 4);
  float*    f_att = (float*)carve((size_t)NQn * 160 * 4);

  // --------- weight pre-pack into WMMA B-fragment layout ---------
  auto packW = [&](const float* W, int K, int CO) -> const _Float16* {
    int ntile = (CO + 15) >> 4;
    _Float16* dst = (_Float16*)carve((size_t)ntile * 512 * 2);
    k_packB<<<divUp(ntile * 512, 256), 256, 0, stream>>>(W, dst, K, CO);
    return dst;
  };

  const float* q0 = qkvs;
  const float* k0 = qkvs + (size_t)NN * XST;
  const float* v0 = qkvs + (size_t)2 * NN * XST;
  const float* si = qkvs + (size_t)3 * NN * XST;

  // --------- edge geometry (transformer, R_MAX=1.5) ---------
  k_geom_edges<<<divUp(NE, 256), 256, 0, stream>>>(pos, esrc, edst, 1.5f,
                                                   sh_e, rb_e, w_e, NE);

  auto runTransformer = [&](const float* const* T, float* featOut) {
    const _Float16 *pQ[4], *pKm[4], *pV[4], *pS[4], *pM1[4], *pM2[4];
    for (int i = 0; i < 4; ++i) {
      pQ[i]  = packW(T[1] + i * 1024, 32, 32);
      pKm[i] = packW(T[2] + i * 1024, 32, 32);
      pV[i]  = packW(T[3] + i * 1024, 32, 32);
      pS[i]  = packW(T[4] + i * 1024, 32, 32);
      pM1[i] = packW(T[5] + i * 320, Bb, Hh);
      pM2[i] = packW(T[7] + i * 12288, 32, 384);
    }
    k_seed_x<<<divUp(NN * Cc, 256), 256, 0, stream>>>(feature, T[0], xbuf);
    for (int i = 0; i < 4; ++i) {
      k_cmix<32, 4><<<NN / 8, 256, 0, stream>>>(xbuf, pQ[i], pKm[i], pV[i],
                                                pS[i], qkvs, NN);
      k_mlp1<<<divUp(NE / 16, 8), 256, 0, stream>>>(rb_e, pM1[i], T[6] + i * 32,
                                                    hidp, NE);
      k_mlp2<<<divUp((NE / 16) * (384 / 16), 8), 256, 0, stream>>>(
          hidp, pM2[i], T[8] + i * 384, gbuf, NE, 384);
      hipMemsetAsync(mbuf, 0, (size_t)NN * 4, stream);
      hipMemsetAsync(denb, 0, (size_t)NN * 4, stream);
      k_logits<<<NE / 8, 256, 0, stream>>>(q0, k0, gbuf, esrc, edst, logit,
                                           mbuf, NE);
      k_softden<<<divUp(NE, 256), 256, 0, stream>>>(logit, mbuf, edst, ealp,
                                                    denb, NE);
      hipMemsetAsync(agg, 0, (size_t)NN * XST * 4, stream);
      k_aggregate<<<NE / 8, 256, 0, stream>>>(ealp, denb, w_e, gbuf, v0, xbuf,
                                              sh_e, esrc, edst, agg, NE);
      k_ln_x<<<NN / 8, 256, 0, stream>>>(agg, si, xbuf, T[9] + i * 128,
                                         (i > 0) ? 1 : 0, NN);
    }
    hipMemcpyAsync(featOut, xbuf, (size_t)NN * XST * 4,
                   hipMemcpyDeviceToDevice, stream);
  };

  struct GateP { const _Float16 *p1, *p2; const float *b1, *b2; };
  auto packGate = [&](const float* const* F, int G) -> GateP {
    GateP gp;
    gp.p1 = packW(F[1], Bb, Hh);
    gp.p2 = packW(F[3], 32, G);
    gp.b1 = F[2];
    gp.b2 = F[4];
    return gp;
  };
  auto runGate = [&](const GateP& gp, int P, int G) {
    k_mlp1<<<divUp(P / 16, 8), 256, 0, stream>>>(rb_p, gp.p1, gp.b1, hidp, P);
    k_mlp2<<<divUp((P / 16) * (G / 16), 8), 256, 0, stream>>>(hidp, gp.p2,
                                                              gp.b2, gbuf, P, G);
  };

  // =========== feat_a = transformer(params_a) ===========
  runTransformer(TA, feata);

  // =========== positional-weight field over edges (FCUT=2.0) ===========
  const _Float16* pWf_pw = packW(PW[0], 32, 16);
  GateP gp_pw = packGate(PW, 128);
  k_geom_edges<<<divUp(NE, 256), 256, 0, stream>>>(pos, esrc, edst, 2.0f,
                                                   sh_p, rb_p, w_p, NE);
  k_cmix<16, 1><<<NN / 8, 256, 0, stream>>>(feata, pWf_pw, pWf_pw, pWf_pw,
                                            pWf_pw, xW, NN);
  runGate(gp_pw, NE, 128);
  hipMemsetAsync(numf, 0, (size_t)NN * 256 * 4, stream);
  hipMemsetAsync(denf, 0, (size_t)NN * 4, stream);
  k_field_msg<<<NE / 8, 256, 0, stream>>>(0, esrc, edst, xW, gbuf, sh_p, w_p,
                                          numf, denf, NE, 16);
  k_field_fin<<<NN / 8, 256, 0, stream>>>(numf, denf, PW[5], f_pw, NN, 16);
  hipMemsetAsync(hbuf, 0, 64 * 4, stream);
  k_pw_head<<<divUp(NN, 256), 256, 0, stream>>>(f_pw, pos, PW[6], PW[7], hbuf);
  k_qp_init<<<1, 32, 0, stream>>>(hbuf, qp);

  // =========== 7 iterations of query displacement ===========
  const _Float16* pWf_dp = packW(DP[0], 32, 16);
  GateP gp_dp = packGate(DP, 128);
  k_cmix<16, 1><<<NN / 8, 256, 0, stream>>>(feata, pWf_dp, pWf_dp, pWf_dp,
                                            pWf_dp, xW, NN);
  for (int it = 0; it < 7; ++it) {
    k_geom_pairs<<<divUp(NPp, 256), 256, 0, stream>>>(pos, qp, 2.0f, sh_p,
                                                      rb_p, w_p, NPp);
    runGate(gp_dp, NPp, 128);
    hipMemsetAsync(numf, 0, (size_t)NQn * 256 * 4, stream);
    hipMemsetAsync(denf, 0, (size_t)NQn * 4, stream);
    k_field_msg<<<NPp / 8, 256, 0, stream>>>(1, nullptr, nullptr, xW, gbuf,
                                             sh_p, w_p, numf, denf, NPp, 16);
    k_field_fin<<<1, 256, 0, stream>>>(numf, denf, DP[5], f_dsp, NQn, 16);
    k_disp_upd<<<1, 32, 0, stream>>>(f_dsp, DP[6], qp);
  }

  // =========== feat_b = transformer(params_b) ===========
  runTransformer(TB, featb);

  // =========== final fields at refined query points ===========
  k_geom_pairs<<<divUp(NPp, 256), 256, 0, stream>>>(pos, qp, 2.0f, sh_p, rb_p,
                                                    w_p, NPp);
  // qfeat (co=32) -> d_out[0 .. 4096)
  const _Float16* pWf_ft = packW(FT[0], 32, 32);
  GateP gp_ft = packGate(FT, 256);
  k_cmix<32, 1><<<NN / 8, 256, 0, stream>>>(featb, pWf_ft, pWf_ft, pWf_ft,
                                            pWf_ft, xW, NN);
  runGate(gp_ft, NPp, 256);
  hipMemsetAsync(numf, 0, (size_t)NQn * 512 * 4, stream);
  hipMemsetAsync(denf, 0, (size_t)NQn * 4, stream);
  k_field_msg<<<NPp / 8, 256, 0, stream>>>(1, nullptr, nullptr, xW, gbuf, sh_p,
                                           w_p, numf, denf, NPp, 32);
  k_field_fin<<<1, 256, 0, stream>>>(numf, denf, FT[5], d_out, NQn, 32);
  // attention field (co=10)
  const _Float16* pWf_at = packW(AT[0], 32, 10);
  GateP gp_at = packGate(AT, 80);
  k_cmix<10, 1><<<NN / 8, 256, 0, stream>>>(featb, pWf_at, pWf_at, pWf_at,
                                            pWf_at, xW, NN);
  runGate(gp_at, NPp, 80);
  hipMemsetAsync(numf, 0, (size_t)NQn * 160 * 4, stream);
  hipMemsetAsync(denf, 0, (size_t)NQn * 4, stream);
  k_field_msg<<<NPp / 8, 256, 0, stream>>>(1, nullptr, nullptr, xW, gbuf, sh_p,
                                           w_p, numf, denf, NPp, 10);
  k_field_fin<<<1, 256, 0, stream>>>(numf, denf, AT[5], f_att, NQn, 10);
  k_attn_head<<<1, 32, 0, stream>>>(f_att, AT[6], AT[7], d_out + 4096 + 24);
  // qp -> d_out[4096 .. 4120)
  hipMemcpyAsync(d_out + 4096, qp, (size_t)NQn * 3 * 4,
                 hipMemcpyDeviceToDevice, stream);
}
